// ParticlePredictor_4398046511649
// MI455X (gfx1250) — matrix operand order repaired and checked by reading, not compiled, not run on hardware
//
#include <hip/hip_runtime.h>
#include <hip/hip_bf16.h>

typedef __attribute__((ext_vector_type(16))) _Float16 v16h;
typedef __attribute__((ext_vector_type(8)))  _Float16 v8h;
typedef __attribute__((ext_vector_type(8)))  float    v8f;

#define GF_RELU   1
#define GF_TSTORE 2

// ---------------- problem constants ----------------
constexpr int  B  = 128, L = 256, C = 126, D = 128;
constexpr int  H  = 8,  DK = 64, DV = 64, FF = 512, NL = 3;
constexpr long BL = (long)B * L;          // 32768 rows
constexpr int  BC = 32;                   // attention batch chunk (scores stay in L2)
constexpr int  KC = 2048;                 // split-K chunk for head GEMM
constexpr int  NKC = (L * D) / KC;        // 16 chunks
constexpr int  MT = 128;                  // macro-tile rows (4 waves x 32)

// ---------------- generic WMMA GEMM (B always N x K "transposed" storage) ----
struct GemmP {
  const _Float16* A; const _Float16* Bm; const float* bias;
  float* Cf; _Float16* Ch;
  int M, N, K, lda, ldb, ldc;
  long sAo, sAi, sBo, sBi, sCo, sCi, sBias;
  long tstride;                // TSTORE: stride per row-block
  int tld;                     // TSTORE: rows per block (=L)
  int innerN, tilesN, flags;
};

__device__ inline v16h load_a_frag(const _Float16* A, int lda, int m0, int k0, int lane) {
  // 16x32 f16 A fragment (CDNA5 ISA 7.12.2): per lane two contiguous 16B chunks
  int m  = m0 + (lane & 15);
  int kb = k0 + ((lane >> 4) << 3);
  const _Float16* p = A + (long)m * lda + kb;
  v8h lo = *(const v8h*)p;
  v8h hi = *(const v8h*)(p + 16);
  v16h r;
#pragma unroll
  for (int i = 0; i < 8; ++i) { r[i] = lo[i]; r[i + 8] = hi[i]; }
  return r;
}

__device__ inline v16h lds_b_frag(const _Float16* row, const _Float16* row16) {
  // row points at K-halfs 8h..8h+7 of column n; hi chunk at +16 (k=16+8h..), row16 unused
  (void)row16;
  v8h lo = *(const v8h*)row;
  v8h hi = *(const v8h*)(row + 16);
  v16h r;
#pragma unroll
  for (int i = 0; i < 8; ++i) { r[i] = lo[i]; r[i + 8] = hi[i]; }
  return r;
}

__global__ __launch_bounds__(128) void wmma_gemm(GemmP p) {
  // double-buffered B panel: 32 cols x 32 k-halfs, rows padded to 40 halfs (80B = 20 banks)
  __shared__ _Float16 sh[2][32][40];
  const int tid  = threadIdx.x;
  const int lane = tid & 31;
  const int wave = tid >> 5;
  const int tile = blockIdx.x;
  const int tm = (tile / p.tilesN) * MT + wave * 32;
  const int tn = (tile % p.tilesN) * 32;
  const int z  = blockIdx.y;
  const int zo = z / p.innerN, zi = z % p.innerN;
  const _Float16* A  = p.A  + (long)zo * p.sAo + (long)zi * p.sAi;
  const _Float16* Bm = p.Bm + (long)zo * p.sBo + (long)zi * p.sBi;

  // async-copy mapping: 128 threads move the 2KB B panel (32 rows x 64B)
  const int  crow = tid >> 2;
  const int  ccol = (tid & 3) * 8;
  const bool bval = (tn + crow) < p.N;
  const _Float16* bsrc = Bm + (long)(tn + crow) * p.ldb + ccol;
  const unsigned lds0 = (unsigned)(uintptr_t)&sh[0][crow][ccol]; // low 32b of flat = LDS offset
  const unsigned lds1 = (unsigned)(uintptr_t)&sh[1][crow][ccol];

  const int nk = p.K >> 5;
  if (bval)
    asm volatile("global_load_async_to_lds_b128 %0, %1, off" :: "v"(lds0), "v"(bsrc) : "memory");
  asm volatile("s_wait_asynccnt 0" ::: "memory");
  __syncthreads();

  v8f acc00 = {}, acc01 = {}, acc10 = {}, acc11 = {};
  for (int ks = 0; ks < nk; ++ks) {
    const int cur = ks & 1;
    if (ks + 1 < nk && bval) {
      const _Float16* nsrc = bsrc + ((ks + 1) << 5);
      asm volatile("global_load_async_to_lds_b128 %0, %1, off"
                   :: "v"(cur ? lds0 : lds1), "v"(nsrc) : "memory");
    }
    const int k0 = ks << 5;
    v16h a0 = load_a_frag(A, p.lda, tm,      k0, lane);
    v16h a1 = load_a_frag(A, p.lda, tm + 16, k0, lane);
    const _Float16* r0 = &sh[cur][(lane & 15)     ][(lane >> 4) << 3];
    const _Float16* r1 = &sh[cur][(lane & 15) + 16][(lane >> 4) << 3];
    v16h b0 = lds_b_frag(r0, r0);
    v16h b1 = lds_b_frag(r1, r1);
    acc00 = __builtin_amdgcn_wmma_f32_16x16x32_f16(false, a0, false, b0, (short)0, acc00, false, false);
    acc01 = __builtin_amdgcn_wmma_f32_16x16x32_f16(false, a0, false, b1, (short)0, acc01, false, false);
    acc10 = __builtin_amdgcn_wmma_f32_16x16x32_f16(false, a1, false, b0, (short)0, acc10, false, false);
    acc11 = __builtin_amdgcn_wmma_f32_16x16x32_f16(false, a1, false, b1, (short)0, acc11, false, false);
    if (ks + 1 < nk)
      asm volatile("s_wait_asynccnt 0" ::: "memory");
    __syncthreads();
  }

  float*    Cf = p.Cf ? p.Cf + (long)zo * p.sCo + (long)zi * p.sCi : nullptr;
  _Float16* Ch = p.Ch ? p.Ch + (long)zo * p.sCo + (long)zi * p.sCi : nullptr;
  const float* bias = p.bias ? p.bias + (long)zo * p.sBias : nullptr;
  const int ncol = tn + (lane & 15);
  const int mrow = tm + ((lane >> 4) << 3);
  const v8f accs[2][2] = {{acc00, acc01}, {acc10, acc11}};
#pragma unroll
  for (int jm = 0; jm < 2; ++jm) {
#pragma unroll
    for (int jn = 0; jn < 2; ++jn) {
      int nn = ncol + jn * 16;
      if (nn >= p.N) continue;
      float bv = bias ? bias[nn] : 0.0f;
      v8f a = accs[jm][jn];
#pragma unroll
      for (int r = 0; r < 8; ++r) {
        int mm = mrow + jm * 16 + r;
        if (mm >= p.M) continue;
        float v = a[r] + bv;
        if ((p.flags & GF_RELU) && v < 0.0f) v = 0.0f;
        if (p.flags & GF_TSTORE) {
          long idx = (long)(mm / p.tld) * p.tstride + (long)nn * p.tld + (mm % p.tld);
          Ch[idx] = (_Float16)v;
        } else {
          long idx = (long)mm * p.ldc + nn;
          if (Cf) Cf[idx] = v;
          if (Ch) Ch[idx] = (_Float16)v;
        }
      }
    }
  }
}

// ---------------- wave reductions ----------------
__device__ inline float wave_sum(float v) {
#pragma unroll
  for (int o = 16; o > 0; o >>= 1) v += __shfl_xor(v, o, 32);
  return v;
}
__device__ inline float wave_max(float v) {
#pragma unroll
  for (int o = 16; o > 0; o >>= 1) v = fmaxf(v, __shfl_xor(v, o, 32));
  return v;
}

// ---------------- Time2Vector + concat ----------------
__global__ __launch_bounds__(32) void time2vec(const float* __restrict__ x,
    const float* __restrict__ wl, const float* __restrict__ bl,
    const float* __restrict__ wp, const float* __restrict__ bp,
    float* __restrict__ h, _Float16* __restrict__ hh) {
  long row = blockIdx.x;            // b*L + l
  int  l    = (int)(row % L);
  int  lane = threadIdx.x;
  const float* xr = x + row * C;
  float s = 0.0f;
  for (int c = lane; c < C; c += 32) s += xr[c];
  float t = wave_sum(s) * (1.0f / (float)C);
  float c0 = wl[l] * t + bl[l];
  float c1 = sinf(wp[l] * t + bp[l]);
  float* hr = h + row * D; _Float16* hhr = hh + row * D;
  for (int d = lane; d < D; d += 32) {
    float v = (d == 0) ? c0 : (d == 1) ? c1 : xr[d - 2];
    hr[d] = v; hhr[d] = (_Float16)v;
  }
}

// ---------------- softmax over L=256 (in-place, f16) ----------------
__global__ __launch_bounds__(32) void softmax256(_Float16* __restrict__ s, float scale) {
  long row = blockIdx.x;
  _Float16* pr = s + row * (long)L;
  int lane = threadIdx.x;
  v8h v = *(const v8h*)(pr + lane * 8);
  float f[8]; float mx = -3.0e30f;
#pragma unroll
  for (int i = 0; i < 8; ++i) { f[i] = (float)v[i] * scale; mx = fmaxf(mx, f[i]); }
  mx = wave_max(mx);
  float sum = 0.0f;
#pragma unroll
  for (int i = 0; i < 8; ++i) { f[i] = __expf(f[i] - mx); sum += f[i]; }
  sum = wave_sum(sum);
  float inv = 1.0f / sum;
  v8h o;
#pragma unroll
  for (int i = 0; i < 8; ++i) o[i] = (_Float16)(f[i] * inv);
  *(v8h*)(pr + lane * 8) = o;
}

// ---------------- residual + LayerNorm over D=128 ----------------
__global__ __launch_bounds__(32) void resid_ln(const float* __restrict__ h,
    const float* __restrict__ y, const float* __restrict__ g,
    const float* __restrict__ be, float* __restrict__ outF,
    _Float16* __restrict__ outH) {
  long row = blockIdx.x;
  int lane = threadIdx.x;
  const float* hr = h + row * D; const float* yr = y + row * D;
  float v[4]; float s = 0.0f, sq = 0.0f;
#pragma unroll
  for (int i = 0; i < 4; ++i) {
    int d = lane * 4 + i;
    float t = hr[d] + yr[d];
    v[i] = t; s += t; sq += t * t;
  }
  s = wave_sum(s); sq = wave_sum(sq);
  float mean = s * (1.0f / D);
  float var  = sq * (1.0f / D) - mean * mean;
  float inv  = rsqrtf(var + 1e-6f);
#pragma unroll
  for (int i = 0; i < 4; ++i) {
    int d = lane * 4 + i;
    float o = g[d] * (v[i] - mean) * inv + be[d];
    if (outF) outF[row * D + d] = o;
    if (outH) outH[row * D + d] = (_Float16)o;
  }
}

// ---------------- f32 [nb][R][Cc] -> f16 [nb][Cc][R] transpose-convert -------
__global__ void convT_f16(const float* __restrict__ src, _Float16* __restrict__ dst,
                          int R, int Cc, int nb) {
  long total = (long)nb * R * Cc;
  long i = (long)blockIdx.x * blockDim.x + threadIdx.x;
  long stride = (long)gridDim.x * blockDim.x;
  for (; i < total; i += stride) {
    long b = i / ((long)R * Cc); long rem = i % ((long)R * Cc);
    int r = (int)(rem / Cc); int c = (int)(rem % Cc);
    dst[b * (long)R * Cc + (long)c * R + r] = (_Float16)src[i];
  }
}

// ---------------- split-K reduce + bias + relu -> f16 ----------------
__global__ void reduce_bias_relu(const float* __restrict__ part, const float* __restrict__ bias,
                                 _Float16* __restrict__ out, int nchunk, int total, int ncols) {
  int i = blockIdx.x * blockDim.x + threadIdx.x;
  if (i >= total) return;
  float s = 0.0f;
  for (int c = 0; c < nchunk; ++c) s += part[(long)c * total + i];
  s += bias[i % ncols];
  if (s < 0.0f) s = 0.0f;
  out[i] = (_Float16)s;
}

// ---------------- host orchestration ----------------
extern "C" void kernel_launch(void* const* d_in, const int* in_sizes, int n_in,
                              void* d_out, int out_size, void* d_ws, size_t ws_size,
                              hipStream_t stream) {
  (void)in_sizes; (void)n_in; (void)out_size; (void)ws_size;
  const float* x   = (const float*)d_in[0];
  const float* wl  = (const float*)d_in[1];
  const float* bl  = (const float*)d_in[2];
  const float* wp  = (const float*)d_in[3];
  const float* bp  = (const float*)d_in[4];
  const float* Wq  = (const float*)d_in[5];
  const float* bq  = (const float*)d_in[6];
  const float* Wk  = (const float*)d_in[7];
  const float* bk  = (const float*)d_in[8];
  const float* Wv  = (const float*)d_in[9];
  const float* bv  = (const float*)d_in[10];
  const float* Wo  = (const float*)d_in[11];
  const float* bo  = (const float*)d_in[12];
  const float* g1  = (const float*)d_in[13];
  const float* be1 = (const float*)d_in[14];
  const float* W1  = (const float*)d_in[15];
  const float* b1  = (const float*)d_in[16];
  const float* W2  = (const float*)d_in[17];
  const float* b2  = (const float*)d_in[18];
  const float* g2  = (const float*)d_in[19];
  const float* be2 = (const float*)d_in[20];
  const float* dW1 = (const float*)d_in[21];
  const float* db1 = (const float*)d_in[22];
  const float* dW2 = (const float*)d_in[23];
  const float* db2 = (const float*)d_in[24];

  // ---- scratch carve-out (~310 MB) ----
  char* wsb = (char*)d_ws;
  size_t off = 0;
  auto alloc = [&](size_t bytes) -> void* {
    void* p = wsb + off;
    off += (bytes + 255) & ~(size_t)255;
    return p;
  };
  float*    h_f    = (float*)alloc((size_t)BL * D * 4);
  float*    attn   = (float*)alloc((size_t)BL * D * 4);
  float*    ffout  = (float*)alloc((size_t)BL * D * 4);
  float*    hidP   = (float*)alloc((size_t)NKC * B * 512 * 4);
  _Float16* h_h    = (_Float16*)alloc((size_t)BL * D * 2);
  _Float16* h1_h   = (_Float16*)alloc((size_t)BL * D * 2);
  _Float16* q_h    = (_Float16*)alloc((size_t)H * BL * DK * 2);
  _Float16* k_h    = (_Float16*)alloc((size_t)H * BL * DK * 2);
  _Float16* vT_h   = (_Float16*)alloc((size_t)H * B * DV * L * 2);   // [H][B][DV][L]
  _Float16* s_h    = (_Float16*)alloc((size_t)H * BC * L * L * 2);
  _Float16* o_h    = (_Float16*)alloc((size_t)BL * H * DV * 2);
  _Float16* ff_h   = (_Float16*)alloc((size_t)BL * FF * 2);
  _Float16* WqT_h  = (_Float16*)alloc((size_t)NL * H * DK * D * 2);  // [NL*H][DK][D]
  _Float16* WkT_h  = (_Float16*)alloc((size_t)NL * H * DK * D * 2);
  _Float16* WvT_h  = (_Float16*)alloc((size_t)NL * H * DV * D * 2);
  _Float16* WoT_h  = (_Float16*)alloc((size_t)NL * D * H * DV * 2);  // [NL][D][512]
  _Float16* W1T_h  = (_Float16*)alloc((size_t)NL * FF * D * 2);      // [NL][FF][D]
  _Float16* W2T_h  = (_Float16*)alloc((size_t)NL * D * FF * 2);      // [NL][D][FF]
  _Float16* dW1T_h = (_Float16*)alloc((size_t)512 * L * D * 2);      // [512][32768]
  _Float16* dW2T_h = (_Float16*)alloc((size_t)C * 512 * 2);          // [126][512]
  _Float16* hid_h  = (_Float16*)alloc((size_t)B * 512 * 2);

  auto convT = [&](const float* src, _Float16* dst, int nb, int R, int Cc) {
    long n = (long)nb * R * Cc;
    long blocks = (n + 255) / 256; if (blocks > 8192) blocks = 8192;
    convT_f16<<<dim3((unsigned)blocks), dim3(256), 0, stream>>>(src, dst, R, Cc, nb);
  };
  convT(Wq,  WqT_h,  NL * H, D, DK);
  convT(Wk,  WkT_h,  NL * H, D, DK);
  convT(Wv,  WvT_h,  NL * H, D, DV);
  convT(Wo,  WoT_h,  NL, H * DV, D);
  convT(W1,  W1T_h,  NL, D, FF);
  convT(W2,  W2T_h,  NL, FF, D);
  convT(dW1, dW1T_h, 1, L * D, 512);
  convT(dW2, dW2T_h, 1, 512, C);

  auto gemm = [&](GemmP p, int batch) {
    int tilesM = (p.M + MT - 1) / MT;
    p.tilesN = (p.N + 31) / 32;
    dim3 grid((unsigned)(tilesM * p.tilesN), (unsigned)batch);
    wmma_gemm<<<grid, dim3(128), 0, stream>>>(p);
  };

  // ---- Time2Vector + concat -> h [BL, D] ----
  time2vec<<<dim3((unsigned)BL), dim3(32), 0, stream>>>(x, wl, bl, wp, bp, h_f, h_h);

  // ---- encoder layers ----
  for (int i = 0; i < NL; ++i) {
    // Q,K projections: [BL,D] x WT[DK,D] per head -> [H][BL][DK]
    struct { const _Float16* W; const float* bia; _Float16* out; } qk[2] = {
      { WqT_h + (long)i * H * DK * D, bq + (long)i * H * DK, q_h },
      { WkT_h + (long)i * H * DK * D, bk + (long)i * H * DK, k_h } };
    for (int j = 0; j < 2; ++j) {
      GemmP p{};
      p.A = h_h; p.lda = D; p.sAo = 0; p.sAi = 0;
      p.Bm = qk[j].W; p.ldb = D; p.sBo = (long)DK * D; p.sBi = 0;
      p.bias = qk[j].bia; p.sBias = DK;
      p.Cf = nullptr; p.Ch = qk[j].out; p.ldc = DK;
      p.sCo = BL * DK; p.sCi = 0;
      p.M = (int)BL; p.N = DK; p.K = D; p.innerN = 1; p.flags = 0;
      gemm(p, H);
    }
    { // V projection stored transposed: [H][B][DV][L]
      GemmP p{};
      p.A = h_h; p.lda = D; p.sAo = 0; p.sAi = 0;
      p.Bm = WvT_h + (long)i * H * DV * D; p.ldb = D; p.sBo = (long)DV * D; p.sBi = 0;
      p.bias = bv + (long)i * H * DV; p.sBias = DV;
      p.Cf = nullptr; p.Ch = vT_h; p.ldc = DV;
      p.sCo = (long)B * DV * L; p.sCi = 0;
      p.tld = L; p.tstride = (long)DV * L;
      p.M = (int)BL; p.N = DV; p.K = D; p.innerN = 1; p.flags = GF_TSTORE;
      gemm(p, H);
    }

    // attention, chunked so the 33.5MB f16 score tile stays in the 192MB L2
    for (int c = 0; c < B / BC; ++c) {
      long cb = (long)c * BC * L * DK;
      { // S = Q K^T
        GemmP p{};
        p.A = q_h + cb; p.lda = DK; p.sAo = BL * DK; p.sAi = (long)L * DK;
        p.Bm = k_h + cb; p.ldb = DK; p.sBo = BL * DK; p.sBi = (long)L * DK;
        p.bias = nullptr; p.sBias = 0;
        p.Cf = nullptr; p.Ch = s_h; p.ldc = L;
        p.sCo = (long)BC * L * L; p.sCi = (long)L * L;
        p.M = L; p.N = L; p.K = DK; p.innerN = BC; p.flags = 0;
        gemm(p, H * BC);
      }
      softmax256<<<dim3((unsigned)(H * BC * L)), dim3(32), 0, stream>>>(s_h, 0.125f);
      { // O = att V  (B panel = V^T, contiguous), written in head-concat layout
        GemmP p{};
        p.A = s_h; p.lda = L; p.sAo = (long)BC * L * L; p.sAi = (long)L * L;
        p.Bm = vT_h + (long)c * BC * DV * L; p.ldb = L;
        p.sBo = (long)B * DV * L; p.sBi = (long)DV * L;
        p.bias = nullptr; p.sBias = 0;
        p.Cf = nullptr; p.Ch = o_h + (long)c * BC * L * (H * DV); p.ldc = H * DV;
        p.sCo = DV; p.sCi = (long)L * H * DV;
        p.M = L; p.N = DV; p.K = L; p.innerN = BC; p.flags = 0;
        gemm(p, H * BC);
      }
    }

    { // output projection: [BL, 512] x WoT[D,512] -> attn (f32)
      GemmP p{};
      p.A = o_h; p.lda = H * DV; p.sAo = 0; p.sAi = 0;
      p.Bm = WoT_h + (long)i * D * H * DV; p.ldb = H * DV; p.sBo = 0; p.sBi = 0;
      p.bias = bo + (long)i * D; p.sBias = 0;
      p.Cf = attn; p.Ch = nullptr; p.ldc = D; p.sCo = 0; p.sCi = 0;
      p.M = (int)BL; p.N = D; p.K = H * DV; p.innerN = 1; p.flags = 0;
      gemm(p, 1);
    }

    // h1 = LN(h + attn)  (f16 only; residual 2 uses h, not h1)
    resid_ln<<<dim3((unsigned)BL), dim3(32), 0, stream>>>(
        h_f, attn, g1 + (long)i * D, be1 + (long)i * D, nullptr, h1_h);

    { // FFN up: relu(h1 W1 + b1) -> ff_h
      GemmP p{};
      p.A = h1_h; p.lda = D; p.sAo = 0; p.sAi = 0;
      p.Bm = W1T_h + (long)i * FF * D; p.ldb = D; p.sBo = 0; p.sBi = 0;
      p.bias = b1 + (long)i * FF; p.sBias = 0;
      p.Cf = nullptr; p.Ch = ff_h; p.ldc = FF; p.sCo = 0; p.sCi = 0;
      p.M = (int)BL; p.N = FF; p.K = D; p.innerN = 1; p.flags = GF_RELU;
      gemm(p, 1);
    }
    { // FFN down: ff W2 + b2 -> ffout (f32)
      GemmP p{};
      p.A = ff_h; p.lda = FF; p.sAo = 0; p.sAi = 0;
      p.Bm = W2T_h + (long)i * D * FF; p.ldb = FF; p.sBo = 0; p.sBi = 0;
      p.bias = b2 + (long)i * D; p.sBias = 0;
      p.Cf = ffout; p.Ch = nullptr; p.ldc = D; p.sCo = 0; p.sCi = 0;
      p.M = (int)BL; p.N = D; p.K = FF; p.innerN = 1; p.flags = 0;
      gemm(p, 1);
    }
    // h = LN(h + ff)  (in-place f32 + f16 mirror)
    resid_ln<<<dim3((unsigned)BL), dim3(32), 0, stream>>>(
        h_f, ffout, g2 + (long)i * D, be2 + (long)i * D, h_f, h_h);
  }

  // ---- head dense 1: [B, 32768] x dW1T[512, 32768], deterministic split-K ----
  {
    GemmP p{};
    p.A = h_h; p.lda = L * D; p.sAo = KC; p.sAi = 0;           // k-chunk offset
    p.Bm = dW1T_h; p.ldb = L * D; p.sBo = KC; p.sBi = 0;       // same k offset in N x K layout
    p.bias = nullptr; p.sBias = 0;
    p.Cf = hidP; p.Ch = nullptr; p.ldc = 512;
    p.sCo = (long)B * 512; p.sCi = 0;                          // per-chunk partials
    p.M = B; p.N = 512; p.K = KC; p.innerN = 1; p.flags = 0;
    gemm(p, NKC);
  }
  reduce_bias_relu<<<dim3((B * 512 + 255) / 256), dim3(256), 0, stream>>>(
      hidP, db1, hid_h, NKC, B * 512, 512);

  // ---- head dense 2: [B, 512] x dW2T[126, 512] -> d_out (f32) ----
  {
    GemmP p{};
    p.A = hid_h; p.lda = 512; p.sAo = 0; p.sAi = 0;
    p.Bm = dW2T_h; p.ldb = 512; p.sBo = 0; p.sBi = 0;
    p.bias = db2; p.sBias = 0;
    p.Cf = (float*)d_out; p.Ch = nullptr; p.ldc = C; p.sCo = 0; p.sCi = 0;
    p.M = B; p.N = C; p.K = 512; p.innerN = 1; p.flags = 0;
    gemm(p, 1);
  }
}
